// FullyConnectedSteerableGeometricProductLayer_36163624633145
// MI455X (gfx1250) — matrix operand order repaired and checked by reading, not compiled, not run on hardware
//
#include <hip/hip_runtime.h>
#include <hip/hip_bf16.h>
#include <math.h>

#define AS3 __attribute__((address_space(3)))

// ---------------------------------------------------------------------------
// Types
// ---------------------------------------------------------------------------
typedef __attribute__((ext_vector_type(16))) __bf16 v16bf;
typedef __attribute__((ext_vector_type(8)))  float  v8f;
typedef __attribute__((ext_vector_type(4)))  unsigned int u32x4;
typedef __attribute__((ext_vector_type(4)))  unsigned int v4u;
typedef __attribute__((ext_vector_type(8)))  int v8i;
typedef __attribute__((ext_vector_type(4)))  int v4i;

// ---------------------------------------------------------------------------
// Problem constants
// ---------------------------------------------------------------------------
constexpr int B_   = 1024;   // batch
constexpr int NIN  = 256;    // input channels
constexpr int NOUT = 256;    // output channels
constexpr int DMV  = 8;      // multivector components (Cl(3,0))
constexpr int K2   = 2304;   // NIN*8 (geometric product) + NIN (left linear)
constexpr float EPSF = 1e-6f;

// GEMM2 LDS staging geometry (dynamic LDS, base offset 0)
constexpr int CHUNK_K   = 64;                    // K elements per staged chunk
constexpr int A_ROWS    = 128;                   // M rows staged per block
constexpr int B_ROWS    = 256;                   // N cols staged per block
constexpr int A_BYTES   = A_ROWS * CHUNK_K * 2;  // 16 KB
constexpr int B_BYTES   = B_ROWS * CHUNK_K * 2;  // 32 KB
constexpr unsigned LDS_A0    = 0;
constexpr unsigned LDS_B0    = 2 * A_BYTES;      // 32768
constexpr unsigned LDS_TOTAL = 2 * A_BYTES + 2 * B_BYTES;  // 96 KB

// ---------------------------------------------------------------------------
// Compile-time Clifford algebra tables (mirrors the Python _build_algebra)
// ---------------------------------------------------------------------------
constexpr int CMASK[8] = {0, 1, 2, 4, 3, 5, 6, 7};  // comp -> mask
constexpr int M2C[8]   = {0, 1, 2, 4, 3, 5, 6, 7};  // mask -> comp
constexpr int GRADE[8] = {0, 1, 1, 1, 2, 2, 2, 3};  // comp -> grade

constexpr int cpopc(int v) { int c = 0; while (v) { c += v & 1; v >>= 1; } return c; }
constexpr int csign(int am, int bm) {
    int a = am >> 1, s = 0;
    while (a) { s += cpopc(a & bm); a >>= 1; }
    return (s & 1) ? -1 : 1;
}
constexpr bool path_exists(int gi, int gj, int gk) {
    for (int a = 0; a < 8; a++)
        for (int b = 0; b < 8; b++)
            if (GRADE[a] == gi && GRADE[b] == gk &&
                GRADE[M2C[CMASK[a] ^ CMASK[b]]] == gj)
                return true;
    return false;
}
constexpr int path_id(int gi, int gj, int gk) {
    int id = 0;
    for (int i = 0; i < 4; i++)
        for (int j = 0; j < 4; j++)
            for (int k = 0; k < 4; k++) {
                if (i == gi && j == gj && k == gk) return id;
                if (path_exists(i, j, k)) id++;
            }
    return -1;
}
struct Tables { int kc[8][8]; float sg[8][8]; int pid[8][8]; };
constexpr Tables make_tables() {
    Tables t{};
    for (int j = 0; j < 8; j++)
        for (int i = 0; i < 8; i++) {
            int mi = CMASK[i], mj = CMASK[j], mk = mi ^ mj;
            int kc = M2C[mk];
            t.kc[j][i]  = kc;
            t.sg[j][i]  = (float)csign(mi, mk);
            t.pid[j][i] = path_id(GRADE[i], GRADE[j], GRADE[kc]);
        }
    return t;
}
constexpr Tables TBL = make_tables();

// ---------------------------------------------------------------------------
// Workspace layout (bytes)
// ---------------------------------------------------------------------------
constexpr size_t OFF_XR   = 0;                                   // f32 [B][NIN][8]
constexpr size_t OFF_XBF  = OFF_XR  + (size_t)B_ * NIN * 8 * 4;  // bf16 [8][B][NIN]
constexpr size_t OFF_WRT  = OFF_XBF + (size_t)8 * B_ * NIN * 2;  // bf16 [4][NIN][NIN]
constexpr size_t OFF_U    = OFF_WRT + (size_t)4 * NIN * NIN * 2; // bf16 [8][B][K2]
constexpr size_t OFF_WEXP = OFF_U   + (size_t)8 * B_ * K2 * 2;   // bf16 [8][NOUT][K2]

// ---------------------------------------------------------------------------
// Global-memory WMMA fragment loaders (A 16x32 / B 32x16, ISA 7.12.2)
// ---------------------------------------------------------------------------
__device__ inline v16bf ldA(const __bf16* __restrict__ base, int ld, int row16,
                            int k0, int lane) {
    const __bf16* p = base + (size_t)(row16 + (lane & 15)) * ld
                           + (k0 + ((lane >> 4) << 3));
    union { v16bf v; u32x4 q[2]; } u;
    u.q[0] = *reinterpret_cast<const u32x4*>(p);
    u.q[1] = *reinterpret_cast<const u32x4*>(p + 16);
    return u.v;
}
__device__ inline v16bf ldB(const __bf16* __restrict__ base, int ld, int col16,
                            int k0, int lane) {
    const __bf16* p = base + (size_t)(col16 + (lane & 15)) * ld
                           + (k0 + ((lane >> 4) << 3));
    union { v16bf v; u32x4 q[2]; } u;
    u.q[0] = *reinterpret_cast<const u32x4*>(p);
    u.q[1] = *reinterpret_cast<const u32x4*>(p + 16);
    return u.v;
}

// ---------------------------------------------------------------------------
// LDS fragment loaders: pointers stay in addrspace(3) (ds_load_b128),
// built from raw byte offsets (dynamic LDS base = 0).
// ---------------------------------------------------------------------------
__device__ inline v16bf ldA_lds(unsigned base, int ld, int row16, int k0, int lane) {
    unsigned off = base +
        (unsigned)(((row16 + (lane & 15)) * ld + k0 + ((lane >> 4) << 3)) * 2);
    const AS3 u32x4* p = (const AS3 u32x4*)off;
    union { v16bf v; u32x4 q[2]; } u;
    u.q[0] = p[0];
    u.q[1] = p[2];                       // +32 bytes (K += 16 elements)
    return u.v;
}
__device__ inline v16bf ldB_lds(unsigned base, int ld, int col16, int k0, int lane) {
    unsigned off = base +
        (unsigned)(((col16 + (lane & 15)) * ld + k0 + ((lane >> 4) << 3)) * 2);
    const AS3 u32x4* p = (const AS3 u32x4*)off;
    union { v16bf v; u32x4 q[2]; } u;
    u.q[0] = p[0];
    u.q[1] = p[2];                       // +32 bytes (K += 16 elements)
    return u.v;
}

// ---------------------------------------------------------------------------
// Compiler fence on an LDS byte offset. The TDM writes LDS behind the
// compiler's back; hiding the offset's provenance (with a memory clobber)
// forces the subsequent ds_loads to actually be performed, while the
// addrspace(3) pointer type keeps them DS ops (not flat).
// ---------------------------------------------------------------------------
__device__ inline unsigned lds_fence_off(unsigned off) {
    asm volatile("" : "+v"(off) : : "memory");
    return off;
}

// ---------------------------------------------------------------------------
// Tensor Data Mover: 2D tile (tileK x tileRows) from row-major global
// (row stride ldk elements, bf16) into LDS at byte offset lds_addr.
// D# layout per CDNA5 ISA 8.3/8.4. 6-arg builtin form (clang-23 / therock).
// ---------------------------------------------------------------------------
#if __has_builtin(__builtin_amdgcn_tensor_load_to_lds)
#define HAVE_TDM 1
__device__ inline void tdm_load_2d(unsigned lds_addr, const void* gaddr,
                                   unsigned tileK, unsigned tileRows,
                                   unsigned ldk) {
    unsigned long long ga = (unsigned long long)(__UINTPTR_TYPE__)gaddr;
    v4u g0;
    g0[0] = 1u;                                        // count=1, user-mode
    g0[1] = lds_addr;                                  // LDS byte address
    g0[2] = (unsigned)(ga & 0xFFFFFFFFu);              // global_addr[31:0]
    g0[3] = (unsigned)((ga >> 32) & 0x01FFFFFFu)       // global_addr[56:32]
          | (2u << 30);                                // type = 2 ("image")
    v8i g1;
    g1[0] = (int)(1u << 16);                           // data_size = 2 bytes
    g1[1] = (int)((ldk & 0xFFFFu) << 16);              // tensor_dim0[15:0] @ b48
    g1[2] = (int)((ldk >> 16) | ((tileRows & 0xFFFFu) << 16)); // dim0 hi / dim1 lo
    g1[3] = (int)(((tileRows >> 16) & 0xFFFFu) | (tileK << 16)); // dim1 hi / tile_dim0
    g1[4] = (int)tileRows;                             // tile_dim1 (tile_dim2=0)
    g1[5] = (int)ldk;                                  // tensor_dim0_stride lo
    g1[6] = 0;                                         // stride hi / dim1_stride lo
    g1[7] = 0;                                         // dim1_stride hi
    v4i z4 = {0, 0, 0, 0};
    v8i z8 = {0, 0, 0, 0, 0, 0, 0, 0};
    __builtin_amdgcn_tensor_load_to_lds(g0, g1, z4, z4, z8, 0);
}
__device__ inline void tdm_wait() { __builtin_amdgcn_s_wait_tensorcnt(0); }
#else
#define HAVE_TDM 0
#endif

// Fallback synchronous staging (global -> VGPR -> LDS), AS3 stores by offset.
__device__ inline void stage_sync(unsigned dst_off, const __bf16* src, int rows,
                                  int ldk, int tid, int nth) {
    const int upr = CHUNK_K / 8;              // 16B units per row
    for (int u = tid; u < rows * upr; u += nth) {
        int r = u / upr, c = (u % upr) * 8;
        AS3 u32x4* d = (AS3 u32x4*)(dst_off + (unsigned)((r * CHUNK_K + c) * 2));
        *d = *reinterpret_cast<const u32x4*>(src + (size_t)r * ldk + c);
    }
}

// ---------------------------------------------------------------------------
// Pack kernels
// ---------------------------------------------------------------------------
__global__ void k_pack_x(const float* __restrict__ x, __bf16* __restrict__ Xbf) {
    int idx = blockIdx.x * blockDim.x + threadIdx.x;   // over B*NIN
    #pragma unroll
    for (int i = 0; i < 8; i++)
        Xbf[(size_t)i * B_ * NIN + idx] = (__bf16)x[(size_t)idx * 8 + i];
}
__global__ void k_pack_wr(const float* __restrict__ wr, __bf16* __restrict__ WrT) {
    int idx = blockIdx.x * blockDim.x + threadIdx.x;   // over NIN*NIN (n*256+m)
    #pragma unroll
    for (int g = 0; g < 4; g++)
        WrT[(size_t)g * NIN * NIN + idx] = (__bf16)wr[(size_t)idx * 4 + g];
}
__global__ void k_pack_gp(const float* __restrict__ gpw,
                          const float* __restrict__ wl,
                          __bf16* __restrict__ Wexp) {
    int idx = blockIdx.x * blockDim.x + threadIdx.x;   // over NOUT*NIN (m*256+n)
    int m = idx >> 8, n = idx & 255;
    float gw[20];
    #pragma unroll
    for (int p = 0; p < 20; p++) gw[p] = gpw[(size_t)idx * 20 + p];
    float wlv[4];
    #pragma unroll
    for (int g = 0; g < 4; g++) wlv[g] = wl[(size_t)idx * 4 + g];
    #pragma unroll
    for (int j = 0; j < 8; j++) {
        __bf16* dst = Wexp + (size_t)j * NOUT * K2 + (size_t)m * K2;
        #pragma unroll
        for (int i = 0; i < 8; i++)
            dst[n * 8 + i] = (__bf16)gw[TBL.pid[j][i]];
        dst[2048 + n] = (__bf16)wlv[GRADE[j]];
    }
}

// ---------------------------------------------------------------------------
// GEMM1: xr[b,n,i] = sum_m x[b,m,i] * w_right[n,m,grade(i)]
// ---------------------------------------------------------------------------
__global__ void k_gemm1(const __bf16* __restrict__ Xbf,
                        const __bf16* __restrict__ WrT,
                        float* __restrict__ xr) {
    const int i = blockIdx.z;
    const int g = (i == 0) ? 0 : (i < 4 ? 1 : (i < 7 ? 2 : 3));
    const int lane = threadIdx.x & 31, w = threadIdx.x >> 5;
    const int Mbase = blockIdx.x * 64 + (w & 1) * 32;
    const int Nbase = blockIdx.y * 128 + (w >> 1) * 32;
    const __bf16* A  = Xbf + (size_t)i * B_ * NIN;
    const __bf16* Bt = WrT + (size_t)g * NIN * NIN;

    v8f acc[2][2] = {};
    for (int k0 = 0; k0 < NIN; k0 += 32) {
        v16bf a0 = ldA(A,  NIN, Mbase,      k0, lane);
        v16bf a1 = ldA(A,  NIN, Mbase + 16, k0, lane);
        v16bf b0 = ldB(Bt, NIN, Nbase,      k0, lane);
        v16bf b1 = ldB(Bt, NIN, Nbase + 16, k0, lane);
        acc[0][0] = __builtin_amdgcn_wmma_f32_16x16x32_bf16(false, a0, false, b0, (short)0, acc[0][0], false, false);
        acc[0][1] = __builtin_amdgcn_wmma_f32_16x16x32_bf16(false, a0, false, b1, (short)0, acc[0][1], false, false);
        acc[1][0] = __builtin_amdgcn_wmma_f32_16x16x32_bf16(false, a1, false, b0, (short)0, acc[1][0], false, false);
        acc[1][1] = __builtin_amdgcn_wmma_f32_16x16x32_bf16(false, a1, false, b1, (short)0, acc[1][1], false, false);
    }
    #pragma unroll
    for (int tm = 0; tm < 2; tm++)
        #pragma unroll
        for (int tn = 0; tn < 2; tn++) {
            int col = Nbase + tn * 16 + (lane & 15);
            #pragma unroll
            for (int r = 0; r < 8; r++) {
                int row = Mbase + tm * 16 + ((lane >> 4) << 3) + r;
                xr[((size_t)row * NIN + col) * DMV + i] = acc[tm][tn][r];
            }
        }
}

// ---------------------------------------------------------------------------
// U build: grade-norm normalize xr, then u[j][b][n*8+i] = sign*x_i*xr_k(i,j),
// plus appended x[b,n,j] columns for the fused left-linear.
// ---------------------------------------------------------------------------
__global__ void k_ubuild(const float* __restrict__ x,
                         const float* __restrict__ xr,
                         const float* __restrict__ norm_a,
                         __bf16* __restrict__ U) {
    int bn = blockIdx.x * blockDim.x + threadIdx.x;  // over B*NIN
    int b = bn >> 8, n = bn & 255;
    float xv[8], rv[8];
    #pragma unroll
    for (int i = 0; i < 8; i++) xv[i] = x[(size_t)bn * 8 + i];
    #pragma unroll
    for (int i = 0; i < 8; i++) rv[i] = xr[(size_t)bn * 8 + i];

    float nrm[4];
    nrm[0] = fabsf(rv[0]);
    nrm[1] = sqrtf(rv[1] * rv[1] + rv[2] * rv[2] + rv[3] * rv[3]);
    nrm[2] = sqrtf(rv[4] * rv[4] + rv[5] * rv[5] + rv[6] * rv[6]);
    nrm[3] = fabsf(rv[7]);
    float den[4];
    #pragma unroll
    for (int g = 0; g < 4; g++) {
        float a = norm_a[n * 4 + g];
        float s = 1.0f / (1.0f + __expf(-a));
        den[g] = s * (nrm[g] - 1.0f) + 1.0f + EPSF;
    }
    #pragma unroll
    for (int i = 0; i < 8; i++) rv[i] /= den[GRADE[i]];

    #pragma unroll
    for (int j = 0; j < 8; j++) {
        union { u32x4 q; __bf16 h[8]; } row;
        #pragma unroll
        for (int i = 0; i < 8; i++)
            row.h[i] = (__bf16)(TBL.sg[j][i] * xv[i] * rv[TBL.kc[j][i]]);
        __bf16* dst = U + (size_t)j * B_ * K2 + (size_t)b * K2;
        *reinterpret_cast<u32x4*>(dst + n * 8) = row.q;
        dst[2048 + n] = (__bf16)xv[j];
    }
}

// ---------------------------------------------------------------------------
// GEMM2 (main): out[b,m,j] = (sum_k U[j][b][k]*Wexp[j][m][k] + bias_j0)/sqrt2
// Block 128M x 256N, 8 waves (2M x 4N), wave tile 64x64.
// K chunked by 64 into double-buffered dynamic LDS via TDM (wave 0 issues
// tensor_load_to_lds; all waves consume via ds_load_b128 by raw offset).
// ---------------------------------------------------------------------------
__global__ void k_gemm2(const __bf16* __restrict__ U,
                        const __bf16* __restrict__ W,
                        const float* __restrict__ bias,
                        float* __restrict__ out) {
    const int j = blockIdx.z;
    const int lane = threadIdx.x & 31, w = threadIdx.x >> 5;
    const int Mblock = blockIdx.x * 128;
    const __bf16* A  = U + (size_t)j * B_ * K2;
    const __bf16* Bt = W + (size_t)j * NOUT * K2;

    constexpr int NCHUNK = K2 / CHUNK_K;   // 36
    v8f acc[4][4] = {};

#if HAVE_TDM
    if (w == 0) {
        tdm_load_2d(LDS_A0, A + (size_t)Mblock * K2, CHUNK_K, A_ROWS, K2);
        tdm_load_2d(LDS_B0, Bt,                      CHUNK_K, B_ROWS, K2);
    }
    for (int kc = 0; kc < NCHUNK; kc++) {
        if (w == 0) tdm_wait();
        __syncthreads();
        if (w == 0 && kc + 1 < NCHUNK) {
            const unsigned nb = (unsigned)((kc + 1) & 1);
            tdm_load_2d(LDS_A0 + nb * A_BYTES,
                        A + (size_t)Mblock * K2 + (kc + 1) * CHUNK_K,
                        CHUNK_K, A_ROWS, K2);
            tdm_load_2d(LDS_B0 + nb * B_BYTES,
                        Bt + (size_t)(kc + 1) * CHUNK_K,
                        CHUNK_K, B_ROWS, K2);
        }
        const unsigned aoff = lds_fence_off(LDS_A0 + (unsigned)(kc & 1) * A_BYTES);
        const unsigned boff = lds_fence_off(LDS_B0 + (unsigned)(kc & 1) * B_BYTES);
        #pragma unroll
        for (int ks = 0; ks < CHUNK_K / 32; ks++) {
            const int k0 = ks * 32;
            v16bf af[4], bfv[4];
            #pragma unroll
            for (int t = 0; t < 4; t++)
                af[t] = ldA_lds(aoff, CHUNK_K, (w & 1) * 64 + t * 16, k0, lane);
            #pragma unroll
            for (int t = 0; t < 4; t++)
                bfv[t] = ldB_lds(boff, CHUNK_K, (w >> 1) * 64 + t * 16, k0, lane);
            #pragma unroll
            for (int tm = 0; tm < 4; tm++)
                #pragma unroll
                for (int tn = 0; tn < 4; tn++)
                    acc[tm][tn] = __builtin_amdgcn_wmma_f32_16x16x32_bf16(
                        false, af[tm], false, bfv[tn], (short)0, acc[tm][tn], false, false);
        }
    }
#else
    for (int kc = 0; kc < NCHUNK; kc++) {
        __syncthreads();
        stage_sync(LDS_A0, A + (size_t)Mblock * K2 + kc * CHUNK_K,
                   A_ROWS, K2, (int)threadIdx.x, 256);
        stage_sync(LDS_B0, Bt + (size_t)kc * CHUNK_K,
                   B_ROWS, K2, (int)threadIdx.x, 256);
        __syncthreads();
        const unsigned aoff = lds_fence_off(LDS_A0);
        const unsigned boff = lds_fence_off(LDS_B0);
        #pragma unroll
        for (int ks = 0; ks < CHUNK_K / 32; ks++) {
            const int k0 = ks * 32;
            v16bf af[4], bfv[4];
            #pragma unroll
            for (int t = 0; t < 4; t++)
                af[t] = ldA_lds(aoff, CHUNK_K, (w & 1) * 64 + t * 16, k0, lane);
            #pragma unroll
            for (int t = 0; t < 4; t++)
                bfv[t] = ldB_lds(boff, CHUNK_K, (w >> 1) * 64 + t * 16, k0, lane);
            #pragma unroll
            for (int tm = 0; tm < 4; tm++)
                #pragma unroll
                for (int tn = 0; tn < 4; tn++)
                    acc[tm][tn] = __builtin_amdgcn_wmma_f32_16x16x32_bf16(
                        false, af[tm], false, bfv[tn], (short)0, acc[tm][tn], false, false);
        }
    }
#endif

    const int Mbase = Mblock + (w & 1) * 64;
    const int Nbase = (w >> 1) * 64;
    const float inv_s2 = 0.70710678118654752440f;
    #pragma unroll
    for (int tm = 0; tm < 4; tm++)
        #pragma unroll
        for (int tn = 0; tn < 4; tn++) {
            int col = Nbase + tn * 16 + (lane & 15);
            float bb = (j == 0) ? bias[col] : 0.0f;
            #pragma unroll
            for (int r = 0; r < 8; r++) {
                int row = Mbase + tm * 16 + ((lane >> 4) << 3) + r;
                out[((size_t)row * NOUT + col) * DMV + j] = (acc[tm][tn][r] + bb) * inv_s2;
            }
        }
}

// ---------------------------------------------------------------------------
// Launch
// ---------------------------------------------------------------------------
extern "C" void kernel_launch(void* const* d_in, const int* in_sizes, int n_in,
                              void* d_out, int out_size, void* d_ws, size_t ws_size,
                              hipStream_t stream) {
    (void)in_sizes; (void)n_in; (void)out_size; (void)ws_size;
    const float* x       = (const float*)d_in[0];  // (B, NIN, 8)
    const float* w_right = (const float*)d_in[1];  // (NIN, NIN, 4)
    const float* w_left  = (const float*)d_in[2];  // (NOUT, NIN, 4)
    const float* b_left  = (const float*)d_in[3];  // (1, NOUT, 1)
    const float* norm_a  = (const float*)d_in[4];  // (NIN, 4)
    const float* gp_w    = (const float*)d_in[5];  // (NOUT, NIN, 20)
    float* out = (float*)d_out;                    // (B, NOUT, 8)

    char* ws = (char*)d_ws;
    float*  xr   = (float*)(ws + OFF_XR);
    __bf16* Xbf  = (__bf16*)(ws + OFF_XBF);
    __bf16* WrT  = (__bf16*)(ws + OFF_WRT);
    __bf16* U    = (__bf16*)(ws + OFF_U);
    __bf16* Wexp = (__bf16*)(ws + OFF_WEXP);

    k_pack_x <<<(B_ * NIN) / 256, 256, 0, stream>>>(x, Xbf);
    k_pack_wr<<<(NIN * NIN) / 256, 256, 0, stream>>>(w_right, WrT);
    k_pack_gp<<<(NOUT * NIN) / 256, 256, 0, stream>>>(gp_w, w_left, Wexp);
    k_gemm1  <<<dim3(B_ / 64, NIN / 128, 8), 256, 0, stream>>>(Xbf, WrT, xr);
    k_ubuild <<<(B_ * NIN) / 256, 256, 0, stream>>>(x, xr, norm_a, U);
    // 96 KB dynamic LDS: base offset 0 matches the TDM descriptors' lds_addr.
    k_gemm2  <<<dim3(B_ / 128, 1, 8), 256, LDS_TOTAL, stream>>>(U, Wexp, b_left, out);
}
